// EGTF_67774583931181
// MI455X (gfx1250) — matrix operand order repaired and checked by reading, not compiled, not run on hardware
//
#include <hip/hip_runtime.h>
#include <hip/hip_bf16.h>

#define NN 4096
#define EE 131072
#define DD 256
#define LL 4
#define NEL 2
#define FFD 512
#define GG 128
#define HH 8

typedef __attribute__((ext_vector_type(16))) __bf16 v16bf;
typedef __attribute__((ext_vector_type(8)))  float  v8f;
typedef unsigned short ushort_t;

union BF16x16 { v16bf v; uint4 q[2]; ushort_t s[16]; };

__device__ __forceinline__ ushort_t f2b(float x) {
  union { float f; unsigned u; } c; c.f = x;
  unsigned r = c.u + 0x7FFFu + ((c.u >> 16) & 1u);
  return (ushort_t)(r >> 16);
}
__device__ __forceinline__ float b2f(ushort_t s) {
  union { unsigned u; float f; } c; c.u = ((unsigned)s) << 16; return c.f;
}
__device__ __forceinline__ float silu_f(float x) { return x / (1.f + __expf(-x)); }
__device__ __forceinline__ float sigm_f(float x) { return 1.f / (1.f + __expf(-x)); }

// A fragment: 16x32 bf16, row-major source (ld in elements).
// Lane m=l&15 -> row; K runs: [half*8 .. +7] and [16+half*8 .. +7].
__device__ __forceinline__ v16bf load_a16x32(const ushort_t* src, int ld) {
  unsigned l = threadIdx.x & 31u, half = l >> 4, m = l & 15u;
  BF16x16 u;
  const ushort_t* r = src + (size_t)m * ld;
  u.q[0] = *(const uint4*)(r + half * 8);
  u.q[1] = *(const uint4*)(r + 16 + half * 8);
  return u.v;
}
// B fragment: 32x16 bf16, row-major [K x N] source; lane n=l&15 -> column; K runs: [half*8 .. +7] and [16+half*8 .. +7].
__device__ __forceinline__ v16bf load_b32x16(const ushort_t* src, int ld) {
  unsigned l = threadIdx.x & 31u, half = l >> 4, n = l & 15u;
  BF16x16 u;
#pragma unroll
  for (int s = 0; s < 16; ++s) {
    unsigned k = (s < 8) ? (half * 8 + s) : (16 + half * 8 + (s - 8));
    u.s[s] = src[(size_t)k * ld + n];
  }
  return u.v;
}
__device__ __forceinline__ v8f wmma_bf16(v16bf a, v16bf b, v8f c) {
  return __builtin_amdgcn_wmma_f32_16x16x32_bf16(false, a, false, b, (short)0, c,
                                                 false, false);
}
__device__ __forceinline__ v8f zero8() {
  v8f z = {0.f, 0.f, 0.f, 0.f, 0.f, 0.f, 0.f, 0.f};
  return z;
}

// ---------------- small utility kernels ----------------
__global__ void k_f2b(const float* __restrict__ s, ushort_t* __restrict__ d, size_t n) {
  size_t i = (size_t)blockIdx.x * 256 + threadIdx.x;
  if (i < n) d[i] = f2b(s[i]);
}
__global__ void k_zero(float* __restrict__ p, size_t n) {
  size_t i = (size_t)blockIdx.x * 256 + threadIdx.x;
  if (i < n) p[i] = 0.f;
}
__global__ void k_embed(const int* __restrict__ z, const float* __restrict__ emb,
                        float* __restrict__ h, ushort_t* __restrict__ hb) {
  size_t i = (size_t)blockIdx.x * 256 + threadIdx.x;  // N*D threads
  int n = (int)(i >> 8), d = (int)(i & 255);
  float v = emb[(size_t)z[n] * DD + d];
  h[i] = v;
  hb[i] = f2b(v);
}
__global__ void k_radial(const float* __restrict__ pos, const int* __restrict__ er,
                         const int* __restrict__ ec, float* __restrict__ rad) {
  int e = blockIdx.x * 256 + threadIdx.x;
  int r = er[e], c = ec[e];
  float dx = pos[r * 3 + 0] - pos[c * 3 + 0];
  float dy = pos[r * 3 + 1] - pos[c * 3 + 1];
  float dz = pos[r * 3 + 2] - pos[c * 3 + 2];
  rad[e] = dx * dx + dy * dy + dz * dz;
}
__global__ void k_seg(const float* __restrict__ h, const int* __restrict__ batch,
                      float* __restrict__ g) {
  size_t i = (size_t)blockIdx.x * 256 + threadIdx.x;  // N*D
  int n = (int)(i >> 8), d = (int)(i & 255);
  atomicAdd(&g[(size_t)batch[n] * DD + d], h[i]);
}
__global__ void k_head(const float* __restrict__ g, const float* __restrict__ w1,
                       const float* __restrict__ b1, const float* __restrict__ w2,
                       const float* __restrict__ b2, float* __restrict__ out) {
  int gi = threadIdx.x;
  if (gi >= GG) return;
  const float* gr = g + (size_t)gi * DD;
  float s = 0.f;
  for (int j = 0; j < DD; ++j) {
    float a = b1[j];
    for (int k = 0; k < DD; ++k) a += gr[k] * w1[k * DD + j];
    s += silu_f(a) * w2[j];
  }
  out[gi] = s + b2[0];
}

// ---------------- generic WMMA GEMM: C = act(A[M,K]@W[K,Nout] + bias) ----------------
template <int NCT, int ACT>
__global__ __launch_bounds__(256) void k_gemm(const ushort_t* __restrict__ A,
                                              const ushort_t* __restrict__ W,
                                              const float* __restrict__ bias,
                                              float* __restrict__ Cf,
                                              ushort_t* __restrict__ Cb, int K,
                                              int Nout) {
  const int wave = threadIdx.x >> 5;
  const int rt = wave & 3, ch = wave >> 2;
  const int row0 = blockIdx.x * 64 + rt * 16;
  const int col0 = ch * (NCT * 16);
  const unsigned l = threadIdx.x & 31u, half = l >> 4, lm = l & 15u;
  v8f acc[NCT];
#pragma unroll
  for (int t = 0; t < NCT; ++t) acc[t] = zero8();
  const int ktiles = K >> 5;
  for (int kt = 0; kt < ktiles; ++kt) {
    v16bf a = load_a16x32(A + (size_t)row0 * K + kt * 32, K);
    __builtin_prefetch(W + (size_t)(kt + 1) * 32 * Nout + col0, 0, 0);
#pragma unroll
    for (int t = 0; t < NCT; ++t) {
      v16bf b = load_b32x16(W + (size_t)kt * 32 * Nout + col0 + t * 16, Nout);
      acc[t] = wmma_bf16(a, b, acc[t]);
    }
  }
#pragma unroll
  for (int t = 0; t < NCT; ++t) {
    int n = col0 + t * 16 + lm;
    float bvv = bias ? bias[n] : 0.f;
#pragma unroll
    for (int r = 0; r < 8; ++r) {
      int row = row0 + r + 8 * half;
      float v = acc[t][r] + bvv;
      if (ACT == 1) v = silu_f(v);
      if (Cf) Cf[(size_t)row * Nout + n] = v;
      if (Cb) Cb[(size_t)row * Nout + n] = f2b(v);
    }
  }
}

// ---------------- EGNN edge kernel (fused edge MLP + gate + scatter) ----------------
__global__ __launch_bounds__(256) void k_edge(
    const ushort_t* __restrict__ hb, const float* __restrict__ rad,
    const int* __restrict__ erow, const int* __restrict__ ecol,
    const ushort_t* __restrict__ W1, const float* __restrict__ w1last,
    const float* __restrict__ b1, const ushort_t* __restrict__ W2,
    const float* __restrict__ b2, const float* __restrict__ attw,
    const float* __restrict__ attb, float* __restrict__ agg) {
  __shared__ __align__(16) ushort_t sA[32 * 520];  // [32 edges][h_row(256)||h_col(256)]
  __shared__ __align__(16) ushort_t sM[32 * 264];  // hidden m (bf16)
  __shared__ float sRad[32];
  __shared__ float sGate[32];
  float* sOut = (float*)sA;  // 32 x 260 f32, reuses sA after GEMM1

  const int e0 = blockIdx.x * 32;
  const int tid = threadIdx.x;
  {  // stage gathered A tile
    int r = tid >> 3, part = tid & 7;
    int e = e0 + r;
    int node = (part < 4) ? erow[e] : ecol[e];
    int cb = (part & 3) * 64 + (part >> 2) * 256;
    const uint4* src = (const uint4*)(hb + (size_t)node * DD + (part & 3) * 64);
    uint4* dst = (uint4*)(sA + r * 520 + cb);
#pragma unroll
    for (int i = 0; i < 8; ++i) dst[i] = src[i];
    if (tid < 32) sRad[tid] = rad[e0 + tid];
  }
  __syncthreads();
  const int wave = tid >> 5;
  const int rt = wave & 1, cq = wave >> 1;
  const int col0 = cq * 64;
  const unsigned l = tid & 31u, half = l >> 4, lm = l & 15u;
  {  // GEMM1: [32,512]@[512,256] + radial rank-1 + bias, SiLU -> sM
    v8f acc[4];
#pragma unroll
    for (int t = 0; t < 4; ++t) acc[t] = zero8();
    for (int kt = 0; kt < 16; ++kt) {
      v16bf a = load_a16x32(sA + rt * 16 * 520 + kt * 32, 520);
      __builtin_prefetch(W1 + (size_t)(kt + 1) * 32 * DD + col0, 0, 0);
#pragma unroll
      for (int t = 0; t < 4; ++t) {
        v16bf b = load_b32x16(W1 + (size_t)kt * 32 * DD + col0 + t * 16, DD);
        acc[t] = wmma_bf16(a, b, acc[t]);
      }
    }
#pragma unroll
    for (int t = 0; t < 4; ++t) {
      int n = col0 + t * 16 + lm;
      float bvv = b1[n], wl = w1last[n];
#pragma unroll
      for (int r = 0; r < 8; ++r) {
        int rowt = rt * 16 + r + 8 * half;
        float v = acc[t][r] + bvv + sRad[rowt] * wl;
        sM[rowt * 264 + n] = f2b(silu_f(v));
      }
    }
  }
  __syncthreads();
  {  // GEMM2: [32,256]@[256,256] + bias, SiLU -> sOut (f32)
    v8f acc[4];
#pragma unroll
    for (int t = 0; t < 4; ++t) acc[t] = zero8();
    for (int kt = 0; kt < 8; ++kt) {
      v16bf a = load_a16x32(sM + rt * 16 * 264 + kt * 32, 264);
#pragma unroll
      for (int t = 0; t < 4; ++t) {
        v16bf b = load_b32x16(W2 + (size_t)kt * 32 * DD + col0 + t * 16, DD);
        acc[t] = wmma_bf16(a, b, acc[t]);
      }
    }
#pragma unroll
    for (int t = 0; t < 4; ++t) {
      int n = col0 + t * 16 + lm;
      float bvv = b2[n];
#pragma unroll
      for (int r = 0; r < 8; ++r) {
        int rowt = rt * 16 + r + 8 * half;
        sOut[rowt * 260 + n] = silu_f(acc[t][r] + bvv);
      }
    }
  }
  __syncthreads();
  if (tid < 32) sGate[tid] = attb[0];
  __syncthreads();
  {  // attention gate logits
    int r = tid >> 3, part = tid & 7;
    float p = 0.f;
    const float* srcp = sOut + r * 260 + part * 32;
    const float* wp = attw + part * 32;
#pragma unroll
    for (int c = 0; c < 32; ++c) p += srcp[c] * wp[c];
    atomicAdd(&sGate[r], p);
  }
  __syncthreads();
  {  // gated scatter-add (segment_sum over edges)
    int r = tid >> 3, part = tid & 7;
    float gv = sigm_f(sGate[r]);
    int node = erow[e0 + r];
    float* dst = agg + (size_t)node * DD + part * 32;
    const float* srcp = sOut + r * 260 + part * 32;
#pragma unroll
    for (int c = 0; c < 32; ++c) atomicAdd(dst + c, srcp[c] * gv);
  }
}

// ---------------- EGNN node kernel (fused node MLP + residual) ----------------
__global__ __launch_bounds__(256) void k_node(float* __restrict__ h,
                                              ushort_t* __restrict__ hb,
                                              const float* __restrict__ agg,
                                              const ushort_t* __restrict__ W1,
                                              const float* __restrict__ b1,
                                              const ushort_t* __restrict__ W2,
                                              const float* __restrict__ b2) {
  __shared__ __align__(16) ushort_t sA[32 * 520];
  __shared__ __align__(16) ushort_t sM[32 * 264];
  const int n0 = blockIdx.x * 32;
  const int tid = threadIdx.x;
  {  // stage [h || bf16(agg)]
    int r = tid >> 3, part = tid & 7;
    int grow = n0 + r;
    int cb = (part & 3) * 64 + (part >> 2) * 256;
    if (part < 4) {
      const uint4* src = (const uint4*)(hb + (size_t)grow * DD + (part & 3) * 64);
      uint4* dst = (uint4*)(sA + r * 520 + cb);
#pragma unroll
      for (int i = 0; i < 8; ++i) dst[i] = src[i];
    } else {
      const float* src = agg + (size_t)grow * DD + (part & 3) * 64;
      ushort_t* dst = sA + r * 520 + cb;
#pragma unroll
      for (int i = 0; i < 64; ++i) dst[i] = f2b(src[i]);
    }
  }
  __syncthreads();
  const int wave = tid >> 5;
  const int rt = wave & 1, cq = wave >> 1;
  const int col0 = cq * 64;
  const unsigned l = tid & 31u, half = l >> 4, lm = l & 15u;
  {
    v8f acc[4];
#pragma unroll
    for (int t = 0; t < 4; ++t) acc[t] = zero8();
    for (int kt = 0; kt < 16; ++kt) {
      v16bf a = load_a16x32(sA + rt * 16 * 520 + kt * 32, 520);
#pragma unroll
      for (int t = 0; t < 4; ++t) {
        v16bf b = load_b32x16(W1 + (size_t)kt * 32 * DD + col0 + t * 16, DD);
        acc[t] = wmma_bf16(a, b, acc[t]);
      }
    }
#pragma unroll
    for (int t = 0; t < 4; ++t) {
      int n = col0 + t * 16 + lm;
      float bvv = b1[n];
#pragma unroll
      for (int r = 0; r < 8; ++r) {
        int rowt = rt * 16 + r + 8 * half;
        sM[rowt * 264 + n] = f2b(silu_f(acc[t][r] + bvv));
      }
    }
  }
  __syncthreads();
  {
    v8f acc[4];
#pragma unroll
    for (int t = 0; t < 4; ++t) acc[t] = zero8();
    for (int kt = 0; kt < 8; ++kt) {
      v16bf a = load_a16x32(sM + rt * 16 * 264 + kt * 32, 264);
#pragma unroll
      for (int t = 0; t < 4; ++t) {
        v16bf b = load_b32x16(W2 + (size_t)kt * 32 * DD + col0 + t * 16, DD);
        acc[t] = wmma_bf16(a, b, acc[t]);
      }
    }
#pragma unroll
    for (int t = 0; t < 4; ++t) {
      int n = col0 + t * 16 + lm;
      float bvv = b2[n];
#pragma unroll
      for (int r = 0; r < 8; ++r) {
        int rowt = rt * 16 + r + 8 * half;
        int grow = n0 + rowt;
        float v = h[(size_t)grow * DD + n] + acc[t][r] + bvv;  // residual
        h[(size_t)grow * DD + n] = v;
        hb[(size_t)grow * DD + n] = f2b(v);
      }
    }
  }
}

// ---------------- flash attention (one wave per 16-query tile x head) ----------------
struct AttnS {
  ushort_t KT[32 * 40];  // K^T tile [dim][key]
  float S[16 * 36];      // score tile
  ushort_t P[16 * 40];   // softmaxed probs (bf16)
  float rowM[16];
  float rowL[16];
  float rowA[16];
};

__global__ __launch_bounds__(256) void k_attn(const ushort_t* __restrict__ qb,
                                              const ushort_t* __restrict__ kb,
                                              const ushort_t* __restrict__ vb,
                                              ushort_t* __restrict__ ob) {
  __shared__ __align__(16) AttnS sm[8];
  const int wave = threadIdx.x >> 5;
  AttnS& me = sm[wave];
  const int gw = blockIdx.x * 8 + wave;
  const int head = gw & (HH - 1);
  const int qt = gw >> 3;
  const unsigned l = threadIdx.x & 31u, half = l >> 4, lm = l & 15u;
  const float scale = 0.17677669529663687f;  // 1/sqrt(32)

  v16bf aQ = load_a16x32(qb + (size_t)(qt * 16) * DD + head * 32, DD);
  v8f acc0 = zero8(), acc1 = zero8();
  if (l < 16) { me.rowM[l] = -1e30f; me.rowL[l] = 0.f; }
  __syncthreads();
  for (int k0 = 0; k0 < NN; k0 += 32) {
    {  // stage transposed K tile: lane l supplies key k0+l
      union { uint4 q[4]; ushort_t s[32]; } kr;
      const uint4* src = (const uint4*)(kb + (size_t)(k0 + l) * DD + head * 32);
      kr.q[0] = src[0]; kr.q[1] = src[1]; kr.q[2] = src[2]; kr.q[3] = src[3];
#pragma unroll
      for (int d = 0; d < 32; ++d) me.KT[d * 40 + l] = kr.s[d];
    }
    __syncthreads();
    v16bf bk0 = load_b32x16(me.KT, 40);
    v16bf bk1 = load_b32x16(me.KT + 16, 40);
    v8f s0 = wmma_bf16(aQ, bk0, zero8());
    v8f s1 = wmma_bf16(aQ, bk1, zero8());
#pragma unroll
    for (int r = 0; r < 8; ++r) {
      int rw = r + 8 * half;
      me.S[rw * 36 + lm] = s0[r] * scale;
      me.S[rw * 36 + 16 + lm] = s1[r] * scale;
    }
    __syncthreads();
    if (l < 16) {  // streaming softmax per query row
      float mold = me.rowM[l];
      float mx = mold;
      const float* Sr = me.S + l * 36;
#pragma unroll
      for (int j = 0; j < 32; ++j) mx = fmaxf(mx, Sr[j]);
      float alpha = __expf(mold - mx);
      float sum = me.rowL[l] * alpha;
#pragma unroll
      for (int j = 0; j < 32; ++j) {
        float p = __expf(Sr[j] - mx);
        sum += p;
        me.P[l * 40 + j] = f2b(p);
      }
      me.rowM[l] = mx; me.rowL[l] = sum; me.rowA[l] = alpha;
    }
    __syncthreads();
    {  // rescale accumulators and add P@V
#pragma unroll
      for (int r = 0; r < 8; ++r) {
        float al = me.rowA[r + 8 * half];
        acc0[r] = acc0[r] * al;
        acc1[r] = acc1[r] * al;
      }
      v16bf aP = load_a16x32(me.P, 40);
      v16bf bv0 = load_b32x16(vb + (size_t)k0 * DD + head * 32, DD);
      v16bf bv1 = load_b32x16(vb + (size_t)k0 * DD + head * 32 + 16, DD);
      acc0 = wmma_bf16(aP, bv0, acc0);
      acc1 = wmma_bf16(aP, bv1, acc1);
    }
    __syncthreads();
  }
#pragma unroll
  for (int r = 0; r < 8; ++r) {
    int rw = r + 8 * half;
    float inv = 1.f / me.rowL[rw];
    size_t row = (size_t)(qt * 16 + rw) * DD + head * 32;
    ob[row + lm] = f2b(acc0[r] * inv);
    ob[row + 16 + lm] = f2b(acc1[r] * inv);
  }
}

// ---------------- residual + LayerNorm (wave32 shuffle reduction) ----------------
__global__ __launch_bounds__(256) void k_ln(float* __restrict__ h,
                                            ushort_t* __restrict__ hb,
                                            const float* __restrict__ tmp,
                                            const float* __restrict__ g,
                                            const float* __restrict__ b) {
  const int row = blockIdx.x * 8 + (threadIdx.x >> 5);
  const unsigned l = threadIdx.x & 31u;
  float x[8];
  float mean = 0.f;
  const float* hr = h + (size_t)row * DD;
  const float* tr = tmp + (size_t)row * DD;
#pragma unroll
  for (int i = 0; i < 8; ++i) {
    float v = hr[l + 32 * i] + tr[l + 32 * i];
    x[i] = v;
    mean += v;
  }
#pragma unroll
  for (int off = 16; off > 0; off >>= 1) mean += __shfl_xor(mean, off, 32);
  mean *= (1.f / 256.f);
  float var = 0.f;
#pragma unroll
  for (int i = 0; i < 8; ++i) { float d = x[i] - mean; var += d * d; }
#pragma unroll
  for (int off = 16; off > 0; off >>= 1) var += __shfl_xor(var, off, 32);
  var *= (1.f / 256.f);
  float rs = rsqrtf(var + 1e-5f);
#pragma unroll
  for (int i = 0; i < 8; ++i) {
    int c = l + 32 * i;
    float y = (x[i] - mean) * rs * g[c] + b[c];
    h[(size_t)row * DD + c] = y;
    hb[(size_t)row * DD + c] = f2b(y);
  }
}

extern "C" void kernel_launch(void* const* d_in, const int* in_sizes, int n_in,
                              void* d_out, int out_size, void* d_ws, size_t ws_size,
                              hipStream_t stream) {
  (void)in_sizes; (void)n_in; (void)out_size; (void)ws_size;
  const int* z = (const int*)d_in[0];
  const float* pos = (const float*)d_in[1];
  const int* batch = (const int*)d_in[2];
  const int* eidx = (const int*)d_in[3];
  const float* emb = (const float*)d_in[4];
  const float* ew1 = (const float*)d_in[5];
  const float* eb1 = (const float*)d_in[6];
  const float* ew2 = (const float*)d_in[7];
  const float* eb2 = (const float*)d_in[8];
  const float* nw1 = (const float*)d_in[9];
  const float* nb1 = (const float*)d_in[10];
  const float* nw2 = (const float*)d_in[11];
  const float* nb2 = (const float*)d_in[12];
  const float* aw = (const float*)d_in[13];
  const float* ab = (const float*)d_in[14];
  const float* wq = (const float*)d_in[15];  const float* bq = (const float*)d_in[16];
  const float* wk = (const float*)d_in[17];  const float* bk = (const float*)d_in[18];
  const float* wv = (const float*)d_in[19];  const float* bvp = (const float*)d_in[20];
  const float* wo = (const float*)d_in[21];  const float* bo = (const float*)d_in[22];
  const float* l1g = (const float*)d_in[23]; const float* l1b = (const float*)d_in[24];
  const float* f1w = (const float*)d_in[25]; const float* f1bias = (const float*)d_in[26];
  const float* f2w = (const float*)d_in[27]; const float* f2bias = (const float*)d_in[28];
  const float* l2g = (const float*)d_in[29]; const float* l2b = (const float*)d_in[30];
  const float* fw1 = (const float*)d_in[31]; const float* fb1 = (const float*)d_in[32];
  const float* fw2 = (const float*)d_in[33]; const float* fb2 = (const float*)d_in[34];

  size_t off = 0;
  char* base = (char*)d_ws;
  auto carve = [&](size_t bytes) {
    char* p = base + off;
    off += (bytes + 255) & ~(size_t)255;
    return p;
  };
  float* h = (float*)carve((size_t)NN * DD * 4);
  ushort_t* hb = (ushort_t*)carve((size_t)NN * DD * 2);
  float* agg = (float*)carve((size_t)NN * DD * 4);
  float* rad = (float*)carve((size_t)EE * 4);
  ushort_t* We1 = (ushort_t*)carve((size_t)LL * 512 * DD * 2);
  ushort_t* We2 = (ushort_t*)carve((size_t)LL * DD * DD * 2);
  ushort_t* Wn1 = (ushort_t*)carve((size_t)LL * 512 * DD * 2);
  ushort_t* Wn2 = (ushort_t*)carve((size_t)LL * DD * DD * 2);
  ushort_t* Wqb = (ushort_t*)carve((size_t)NEL * DD * DD * 2);
  ushort_t* Wkb = (ushort_t*)carve((size_t)NEL * DD * DD * 2);
  ushort_t* Wvb = (ushort_t*)carve((size_t)NEL * DD * DD * 2);
  ushort_t* Wob = (ushort_t*)carve((size_t)NEL * DD * DD * 2);
  ushort_t* Wf1 = (ushort_t*)carve((size_t)NEL * DD * FFD * 2);
  ushort_t* Wf2 = (ushort_t*)carve((size_t)NEL * FFD * DD * 2);
  ushort_t* qbuf = (ushort_t*)carve((size_t)NN * DD * 2);
  ushort_t* kbuf = (ushort_t*)carve((size_t)NN * DD * 2);
  ushort_t* vbuf = (ushort_t*)carve((size_t)NN * DD * 2);
  ushort_t* abuf = (ushort_t*)carve((size_t)NN * DD * 2);
  ushort_t* fbuf = (ushort_t*)carve((size_t)NN * FFD * 2);
  float* tmp = (float*)carve((size_t)NN * DD * 4);
  float* gp = (float*)carve((size_t)GG * DD * 4);

  auto cvt = [&](const float* s, ushort_t* d, size_t n) {
    k_f2b<<<(unsigned)((n + 255) / 256), 256, 0, stream>>>(s, d, n);
  };
  for (int i = 0; i < LL; ++i)  // edge W1: skip the fp32 radial row (row 512)
    cvt(ew1 + (size_t)i * 513 * DD, We1 + (size_t)i * 512 * DD, (size_t)512 * DD);
  cvt(ew2, We2, (size_t)LL * DD * DD);
  cvt(nw1, Wn1, (size_t)LL * 512 * DD);
  cvt(nw2, Wn2, (size_t)LL * DD * DD);
  cvt(wq, Wqb, (size_t)NEL * DD * DD);
  cvt(wk, Wkb, (size_t)NEL * DD * DD);
  cvt(wv, Wvb, (size_t)NEL * DD * DD);
  cvt(wo, Wob, (size_t)NEL * DD * DD);
  cvt(f1w, Wf1, (size_t)NEL * DD * FFD);
  cvt(f2w, Wf2, (size_t)NEL * FFD * DD);

  k_embed<<<NN * DD / 256, 256, 0, stream>>>(z, emb, h, hb);
  k_radial<<<EE / 256, 256, 0, stream>>>(pos, eidx, eidx + EE, rad);

  for (int i = 0; i < LL; ++i) {
    k_zero<<<NN * DD / 256, 256, 0, stream>>>(agg, (size_t)NN * DD);
    k_edge<<<EE / 32, 256, 0, stream>>>(
        hb, rad, eidx, eidx + EE, We1 + (size_t)i * 512 * DD,
        ew1 + (size_t)i * 513 * DD + (size_t)512 * DD, eb1 + i * DD,
        We2 + (size_t)i * DD * DD, eb2 + i * DD, aw + i * DD, ab + i, agg);
    k_node<<<NN / 32, 256, 0, stream>>>(h, hb, agg, Wn1 + (size_t)i * 512 * DD,
                                        nb1 + i * DD, Wn2 + (size_t)i * DD * DD,
                                        nb2 + i * DD);
  }

  for (int i = 0; i < NEL; ++i) {
    k_gemm<8, 0><<<NN / 64, 256, 0, stream>>>(hb, Wqb + (size_t)i * DD * DD,
                                              bq + i * DD, nullptr, qbuf, DD, DD);
    k_gemm<8, 0><<<NN / 64, 256, 0, stream>>>(hb, Wkb + (size_t)i * DD * DD,
                                              bk + i * DD, nullptr, kbuf, DD, DD);
    k_gemm<8, 0><<<NN / 64, 256, 0, stream>>>(hb, Wvb + (size_t)i * DD * DD,
                                              bvp + i * DD, nullptr, vbuf, DD, DD);
    k_attn<<<NN * HH / (16 * 8), 256, 0, stream>>>(qbuf, kbuf, vbuf, abuf);
    k_gemm<8, 0><<<NN / 64, 256, 0, stream>>>(abuf, Wob + (size_t)i * DD * DD,
                                              bo + i * DD, tmp, nullptr, DD, DD);
    k_ln<<<NN / 8, 256, 0, stream>>>(h, hb, tmp, l1g + i * DD, l1b + i * DD);
    k_gemm<16, 1><<<NN / 64, 256, 0, stream>>>(hb, Wf1 + (size_t)i * DD * FFD,
                                               f1bias + i * FFD, nullptr, fbuf, DD,
                                               FFD);
    k_gemm<8, 0><<<NN / 64, 256, 0, stream>>>(fbuf, Wf2 + (size_t)i * FFD * DD,
                                              f2bias + i * DD, tmp, nullptr, FFD, DD);
    k_ln<<<NN / 8, 256, 0, stream>>>(h, hb, tmp, l2g + i * DD, l2b + i * DD);
  }

  k_zero<<<(GG * DD) / 256, 256, 0, stream>>>(gp, (size_t)GG * DD);
  k_seg<<<NN * DD / 256, 256, 0, stream>>>(h, batch, gp);
  k_head<<<1, 128, 0, stream>>>(gp, fw1, fb1, fw2, fb2, (float*)d_out);
}
